// MultiHeadISSMScan_67044439491010
// MI455X (gfx1250) — compile-verified, matrix operand order checked by reading, not run on hardware
//
#include <hip/hip_runtime.h>
#include <hip/hip_bf16.h>
#include <math.h>

// ---------------------------------------------------------------------------
// MultiHeadISSMScan for MI455X (gfx1250, wave32, WMMA)
//   B=2, L=2048, K=64, d_model=512, d_inner=1024, H=8, P=128, G=1, d_dist=4
// GEMM operands pre-converted to bf16 (halves repeated HBM traffic);
// hot loop: 10 batched global_load_b128 per K-slice (sched_barrier-fenced),
// then 4 x v_wmma_f32_16x16x32_bf16 sharing one A fragment.
// Scan -> VALU/LDS serial recurrence with prefetch.
// ---------------------------------------------------------------------------

typedef __attribute__((ext_vector_type(16))) __bf16 v16bf;
typedef __attribute__((ext_vector_type(8)))  __bf16 v8bf;
typedef __attribute__((ext_vector_type(8)))  float  v8f;

#define BB   2
#define LL   2048
#define KK   64
#define DMODEL 512
#define DINNER 1024
#define HH   8
#define PP   128
#define NPROJ 2058      // 2*d_inner + 2*G + H
#define CCONV 1026      // d_inner + 2*G

// ---------------------------------------------------------------------------
// fp32 -> bf16 bulk convert (8 elements / thread, b128 in / b128 out)
// ---------------------------------------------------------------------------
__global__ void f32_to_bf16_kernel(const float* __restrict__ in,
                                   __bf16* __restrict__ out, int n8)
{
    const int i = blockIdx.x * blockDim.x + threadIdx.x;
    if (i >= n8) return;
    const float4 f0 = *(const float4*)(in + (size_t)i * 8);
    const float4 f1 = *(const float4*)(in + (size_t)i * 8 + 4);
    v8bf r;
    r[0] = (__bf16)f0.x; r[1] = (__bf16)f0.y; r[2] = (__bf16)f0.z; r[3] = (__bf16)f0.w;
    r[4] = (__bf16)f1.x; r[5] = (__bf16)f1.y; r[6] = (__bf16)f1.z; r[7] = (__bf16)f1.w;
    *(v8bf*)(out + (size_t)i * 8) = r;
}

// ---------------------------------------------------------------------------
// C[M,N] = A[M,Kd] * W[N,Kd]^T  (bf16 in, WMMA f32-acc, fp32 out)
// block = 128 threads (4 waves). Wave w owns rows [bx*64+w*16, +16) and
// columns [by*64, +64): 4 WMMA tiles sharing one A fragment per K-slice.
// ISA 7.12.2 (16-bit A, 16x32): lane's 16 elements are two contiguous
// 8-element runs at +0 and +16 (K-halves selected by lane>=16 via hi8).
// Requires: M % 64 == 0, Kd % 32 == 0. N may be ragged (store-guarded,
// loads clamp the column index -> branchless).
// grid = (M/64, ceil(N/64))
// ---------------------------------------------------------------------------
__global__ void __launch_bounds__(128, 1)
gemm_wmma_bf16(const __bf16* __restrict__ A,
               const __bf16* __restrict__ W,
               float* __restrict__ C,
               int M, int N, int Kd)
{
    const int lane = threadIdx.x & 31;
    const int wave = threadIdx.x >> 5;
    const int hi8  = ((lane >> 4) & 1) * 8;
    const int m0   = blockIdx.x * 64 + wave * 16;
    const int n0   = blockIdx.y * 64;

    const int row  = m0 + (lane & 15);
    const __bf16* __restrict__ arow = A + (size_t)row * Kd + hi8;

    const __bf16* __restrict__ wcol0;
    const __bf16* __restrict__ wcol1;
    const __bf16* __restrict__ wcol2;
    const __bf16* __restrict__ wcol3;
    {
        int c0 = n0 +  0 + (lane & 15); c0 = (c0 < N) ? c0 : (N - 1);
        int c1 = n0 + 16 + (lane & 15); c1 = (c1 < N) ? c1 : (N - 1);
        int c2 = n0 + 32 + (lane & 15); c2 = (c2 < N) ? c2 : (N - 1);
        int c3 = n0 + 48 + (lane & 15); c3 = (c3 < N) ? c3 : (N - 1);
        wcol0 = W + (size_t)c0 * Kd + hi8;
        wcol1 = W + (size_t)c1 * Kd + hi8;
        wcol2 = W + (size_t)c2 * Kd + hi8;
        wcol3 = W + (size_t)c3 * Kd + hi8;
    }

    v8f acc0 = {}, acc1 = {}, acc2 = {}, acc3 = {};
    for (int kk = 0; kk < Kd; kk += 32) {
        // ---- load phase: 10 independent global_load_b128 ----
        const v8bf aL  = *(const v8bf*)(arow  + kk);
        const v8bf aH  = *(const v8bf*)(arow  + kk + 16);
        const v8bf b0L = *(const v8bf*)(wcol0 + kk);
        const v8bf b0H = *(const v8bf*)(wcol0 + kk + 16);
        const v8bf b1L = *(const v8bf*)(wcol1 + kk);
        const v8bf b1H = *(const v8bf*)(wcol1 + kk + 16);
        const v8bf b2L = *(const v8bf*)(wcol2 + kk);
        const v8bf b2H = *(const v8bf*)(wcol2 + kk + 16);
        const v8bf b3L = *(const v8bf*)(wcol3 + kk);
        const v8bf b3H = *(const v8bf*)(wcol3 + kk + 16);
        // hard scheduling fence: issue everything above before anything below
        __builtin_amdgcn_sched_barrier(0);

        // ---- assemble fragments (register concat, no ALU) ----
        const v16bf a  = __builtin_shufflevector(aL,  aH,  0,1,2,3,4,5,6,7,8,9,10,11,12,13,14,15);
        const v16bf b0 = __builtin_shufflevector(b0L, b0H, 0,1,2,3,4,5,6,7,8,9,10,11,12,13,14,15);
        const v16bf b1 = __builtin_shufflevector(b1L, b1H, 0,1,2,3,4,5,6,7,8,9,10,11,12,13,14,15);
        const v16bf b2 = __builtin_shufflevector(b2L, b2H, 0,1,2,3,4,5,6,7,8,9,10,11,12,13,14,15);
        const v16bf b3 = __builtin_shufflevector(b3L, b3H, 0,1,2,3,4,5,6,7,8,9,10,11,12,13,14,15);

        // ---- 4 WMMAs sharing the A fragment ----
        acc0 = __builtin_amdgcn_wmma_f32_16x16x32_bf16(false, a, false, b0, (short)0, acc0, false, false);
        acc1 = __builtin_amdgcn_wmma_f32_16x16x32_bf16(false, a, false, b1, (short)0, acc1, false, false);
        acc2 = __builtin_amdgcn_wmma_f32_16x16x32_bf16(false, a, false, b2, (short)0, acc2, false, false);
        acc3 = __builtin_amdgcn_wmma_f32_16x16x32_bf16(false, a, false, b3, (short)0, acc3, false, false);
    }

    // C/D layout: VGPR v -> row m0 + v + (lane>=16 ? 8 : 0), col = lane&15.
#pragma unroll
    for (int j = 0; j < 4; ++j) {
        const v8f acc = (j == 0) ? acc0 : (j == 1) ? acc1 : (j == 2) ? acc2 : acc3;
        const int col = n0 + j * 16 + (lane & 15);
        if (col < N) {
#pragma unroll
            for (int v = 0; v < 8; ++v) {
                const int m = m0 + v + hi8;
                C[(size_t)m * N + col] = acc[v];
            }
        }
    }
}

// ---------------------------------------------------------------------------
// Depthwise conv (kernel 7, pad 3) + SiLU, forward & backward weight sets.
// One thread per (b, l, c) element.
// ---------------------------------------------------------------------------
__global__ void dwconv_silu_kernel(const float* __restrict__ zxbcdt,
                                   const float* __restrict__ wf, const float* __restrict__ bf,
                                   const float* __restrict__ wb, const float* __restrict__ bb,
                                   float* __restrict__ xbc_f, float* __restrict__ xbc_b)
{
    const int idx = blockIdx.x * blockDim.x + threadIdx.x;
    const int total = BB * LL * CCONV;
    if (idx >= total) return;
    const int c  = idx % CCONV;
    const int bl = idx / CCONV;
    const int l  = bl % LL;
    const int b  = bl / LL;

    float af = bf[c];
    float ab = bb[c];
#pragma unroll
    for (int j = 0; j < 7; ++j) {
        const int ll = l + j - 3;
        if (ll >= 0 && ll < LL) {
            const float v = zxbcdt[((size_t)b * LL + ll) * NPROJ + DINNER + c];
            af += wf[c * 7 + j] * v;
            ab += wb[c * 7 + j] * v;
        }
    }
    af = af * (1.0f / (1.0f + __expf(-af)));   // SiLU
    ab = ab * (1.0f / (1.0f + __expf(-ab)));
    xbc_f[(size_t)bl * CCONV + c] = af;
    xbc_b[(size_t)bl * CCONV + c] = ab;
}

// ---------------------------------------------------------------------------
// dt = softplus(dist @ dt_proj_w.T + dt_head + dt_bias)  -> (B,L,H,K)
// b_base/c_base = dist @ bc_proj_w.T (swapped)            -> (B,L,K)
// One thread per (b,l,k).
// ---------------------------------------------------------------------------
__global__ void dt_base_kernel(const float* __restrict__ dist,
                               const float* __restrict__ zxbcdt,
                               const float* __restrict__ bc_w,   // (2,4)
                               const float* __restrict__ dt_w,   // (8,4)
                               const float* __restrict__ dt_bias,// (8)
                               float* __restrict__ dtb,          // (B,L,H,K)
                               float* __restrict__ bbase,        // (B,L,K)
                               float* __restrict__ cbase)        // (B,L,K)
{
    const int idx = blockIdx.x * blockDim.x + threadIdx.x;
    const int total = BB * LL * KK;
    if (idx >= total) return;
    const int k  = idx % KK;
    const int bl = idx / KK;

    const float4 d = *(const float4*)(dist + (size_t)idx * 4);

    bbase[idx] = d.x * bc_w[0] + d.y * bc_w[1] + d.z * bc_w[2] + d.w * bc_w[3];
    cbase[idx] = d.x * bc_w[4] + d.y * bc_w[5] + d.z * bc_w[6] + d.w * bc_w[7];

#pragma unroll
    for (int h = 0; h < HH; ++h) {
        float s = d.x * dt_w[h * 4 + 0] + d.y * dt_w[h * 4 + 1] +
                  d.z * dt_w[h * 4 + 2] + d.w * dt_w[h * 4 + 3];
        s += zxbcdt[(size_t)bl * NPROJ + 2 * DINNER + 2 + h];  // dt_head
        s += dt_bias[h];
        const float sp = (s > 20.0f) ? s : log1pf(__expf(s));   // softplus
        dtb[((size_t)bl * HH + h) * KK + k] = sp;
    }
}

// ---------------------------------------------------------------------------
// Serial ISSM scan, both directions. grid.x = 2 (dir) * B*H = 32 workgroups.
// 256 threads: thread t owns p = t&127, k-half = t>>7 (32 k values in VGPRs).
// ---------------------------------------------------------------------------
__global__ void issm_scan_kernel(const float* __restrict__ dtb,
                                 const float* __restrict__ bbase,
                                 const float* __restrict__ cbase,
                                 const float* __restrict__ xbc_f,
                                 const float* __restrict__ xbc_b,
                                 const float* __restrict__ h0lin,  // (B*K, d_inner)
                                 const float* __restrict__ A_log,
                                 const float* __restrict__ D_param,
                                 float* __restrict__ y_f,          // (B,L,d_inner)
                                 float* __restrict__ y_b,          // (B,L,d_inner)
                                 float* __restrict__ hF)           // (2,B,H,P,K)
{
    const int wg  = blockIdx.x;
    const int dir = wg >> 4;
    const int b   = (wg >> 3) & 1;
    const int h   = wg & 7;
    const int t   = threadIdx.x;
    const int p   = t & 127;
    const int kh  = t >> 7;          // 0 or 1

    const float Ah = -__expf(A_log[h]);
    const float Dh = D_param[h];

    __shared__ float sDA[KK], sU[KK], sC[KK], sX[PP], sAcc[256];

    // init state from h0:  h0[b,h,p,k] = h0lin[b*K+k, h*P+p]
    float hreg[32];
#pragma unroll
    for (int j = 0; j < 32; ++j) {
        const int k = kh * 32 + j;
        hreg[j] = h0lin[((size_t)b * KK + k) * DINNER + h * PP + p];
    }

    const float* xbc  = dir ? xbc_b : xbc_f;
    float*       yout = dir ? y_b   : y_f;

    for (int s = 0; s < LL; ++s) {
        const int l = dir ? (LL - 1 - s) : s;
        const size_t bl = (size_t)b * LL + l;

        if (t < KK) {
            const int k = t;
            const float dtv = dtb[(bl * HH + h) * KK + k];
            const float bbv = xbc[bl * CCONV + DINNER + 0];
            const float cbv = xbc[bl * CCONV + DINNER + 1];
            sDA[k] = __expf(dtv * Ah);
            sU[k]  = dtv * (bbase[bl * KK + k] + bbv);
            sC[k]  = cbase[bl * KK + k] + cbv;
            // prefetch next step's dt row into cache while we compute
            const int ln = dir ? (l - 1) : (l + 1);
            if (ln >= 0 && ln < LL)
                __builtin_prefetch(&dtb[(((size_t)b * LL + ln) * HH + h) * KK + k], 0, 3);
        } else if (t < KK + PP) {
            const int pp = t - KK;
            sX[pp] = xbc[bl * CCONV + h * PP + pp];
        }
        __syncthreads();

        const float xv = sX[p];
        float acc = 0.0f;
#pragma unroll
        for (int j = 0; j < 32; ++j) {
            const int k = kh * 32 + j;
            hreg[j] = sDA[k] * hreg[j] + xv * sU[k];
            acc += hreg[j] * sC[k];
        }
        sAcc[t] = acc;
        __syncthreads();

        if (t < PP) {
            const float y = sAcc[t] + sAcc[t + PP] + Dh * xv;
            yout[bl * DINNER + h * PP + p] = y;
        }
    }

    // final state: hF[dir][b][h][p][k]
#pragma unroll
    for (int j = 0; j < 32; ++j) {
        const int k = kh * 32 + j;
        hF[((((size_t)dir * BB + b) * HH + h) * PP + p) * KK + k] = hreg[j];
    }
}

// ---------------------------------------------------------------------------
// key path: y = 0.5*(y_f + y_b); xg = y*silu(z); RMSNorm * key_norm_w.
// Emits bf16 directly (GEMM operand). One WG (256 thr) per (b,l) row.
// ---------------------------------------------------------------------------
__global__ void gate_rmsnorm_kernel(const float* __restrict__ y_f,
                                    const float* __restrict__ y_b,
                                    const float* __restrict__ zxbcdt,
                                    const float* __restrict__ key_norm_w,
                                    __bf16* __restrict__ keyn_bf)
{
    const size_t row = blockIdx.x;     // 0 .. B*L-1
    const int t = threadIdx.x;
    __shared__ float red[256];

    float xg[4];
    float ss = 0.0f;
#pragma unroll
    for (int i = 0; i < 4; ++i) {
        const int d = t + i * 256;
        const float y = 0.5f * (y_f[row * DINNER + d] + y_b[row * DINNER + d]);
        const float z = zxbcdt[row * NPROJ + d];
        const float g = y * (z * (1.0f / (1.0f + __expf(-z))));
        xg[i] = g;
        ss += g * g;
    }
    red[t] = ss;
    __syncthreads();
    for (int o = 128; o > 0; o >>= 1) {
        if (t < o) red[t] += red[t + o];
        __syncthreads();
    }
    const float scale = rsqrtf(red[0] * (1.0f / DINNER) + 1e-5f);
#pragma unroll
    for (int i = 0; i < 4; ++i) {
        const int d = t + i * 256;
        keyn_bf[row * DINNER + d] = (__bf16)(xg[i] * scale * key_norm_w[d]);
    }
}

// ---------------------------------------------------------------------------
// query path: q = 0.5*(hF_f + hF_b) transposed to (B,K,d_inner); LayerNorm.
// Emits bf16 directly. One WG per (b,k) row.
// ---------------------------------------------------------------------------
__global__ void qprep_kernel(const float* __restrict__ hF,
                             const float* __restrict__ q_ln_w,
                             const float* __restrict__ q_ln_b,
                             __bf16* __restrict__ qn_bf)
{
    const int row = blockIdx.x;        // 0 .. B*K-1
    const int b = row / KK;
    const int k = row % KK;
    const int t = threadIdx.x;
    __shared__ float red[256];

    float q[4];
    float s = 0.0f;
#pragma unroll
    for (int i = 0; i < 4; ++i) {
        const int d = t + i * 256;
        const int h = d >> 7, p = d & 127;
        const float v0 = hF[((((size_t)0 * BB + b) * HH + h) * PP + p) * KK + k];
        const float v1 = hF[((((size_t)1 * BB + b) * HH + h) * PP + p) * KK + k];
        q[i] = 0.5f * (v0 + v1);
        s += q[i];
    }
    red[t] = s;
    __syncthreads();
    for (int o = 128; o > 0; o >>= 1) { if (t < o) red[t] += red[t + o]; __syncthreads(); }
    const float mu = red[0] * (1.0f / DINNER);
    __syncthreads();

    float vs = 0.0f;
#pragma unroll
    for (int i = 0; i < 4; ++i) { const float d = q[i] - mu; vs += d * d; }
    red[t] = vs;
    __syncthreads();
    for (int o = 128; o > 0; o >>= 1) { if (t < o) red[t] += red[t + o]; __syncthreads(); }
    const float inv = rsqrtf(red[0] * (1.0f / DINNER) + 1e-5f);

#pragma unroll
    for (int i = 0; i < 4; ++i) {
        const int d = t + i * 256;
        qn_bf[(size_t)row * DINNER + d] = (__bf16)((q[i] - mu) * inv * q_ln_w[d] + q_ln_b[d]);
    }
}

// ---------------------------------------------------------------------------
extern "C" void kernel_launch(void* const* d_in, const int* in_sizes, int n_in,
                              void* d_out, int out_size, void* d_ws, size_t ws_size,
                              hipStream_t stream)
{
    const float* in_key      = (const float*)d_in[0];
    const float* in_query    = (const float*)d_in[1];
    const float* dist        = (const float*)d_in[2];
    // d_in[3] key_xyz unused
    const float* key_proj_w  = (const float*)d_in[4];
    const float* key_conv_w  = (const float*)d_in[5];
    const float* key_conv_b  = (const float*)d_in[6];
    const float* key_conv_bw = (const float*)d_in[7];
    const float* key_conv_bb = (const float*)d_in[8];
    const float* query_proj_w= (const float*)d_in[9];
    const float* bc_proj_w   = (const float*)d_in[10];
    const float* dt_proj_w   = (const float*)d_in[11];
    const float* dt_bias     = (const float*)d_in[12];
    const float* A_log       = (const float*)d_in[13];
    const float* D_param     = (const float*)d_in[14];
    const float* out_key_w   = (const float*)d_in[15];
    const float* out_query_w = (const float*)d_in[16];
    const float* key_norm_w  = (const float*)d_in[17];
    const float* q_ln_w      = (const float*)d_in[18];
    const float* q_ln_b      = (const float*)d_in[19];

    float* ws = (float*)d_ws;
    size_t o = 0;
    float* zxbcdt = ws + o; o += (size_t)BB * LL * NPROJ;   // (B,L,2058)
    float* xbc_f  = ws + o; o += (size_t)BB * LL * CCONV;   // (B,L,1026)
    float* xbc_b  = ws + o; o += (size_t)BB * LL * CCONV;
    float* dtb    = ws + o; o += (size_t)BB * LL * HH * KK; // (B,L,H,K)
    float* bbase  = ws + o; o += (size_t)BB * LL * KK;
    float* cbase  = ws + o; o += (size_t)BB * LL * KK;
    float* h0lin  = ws + o; o += (size_t)BB * KK * DINNER;  // (B*K,1024)
    float* y_f    = ws + o; o += (size_t)BB * LL * DINNER;
    float* y_b    = ws + o; o += (size_t)BB * LL * DINNER;
    float* hF     = ws + o; o += (size_t)2 * BB * HH * PP * KK;

    // bf16 GEMM operand pool (element counts are all even)
    __bf16* inkey_bf = (__bf16*)(ws + o); o += (size_t)BB * LL * DMODEL / 2;
    __bf16* keyw_bf  = (__bf16*)(ws + o); o += (size_t)NPROJ * DMODEL / 2;
    __bf16* inq_bf   = (__bf16*)(ws + o); o += (size_t)BB * KK * DMODEL / 2;
    __bf16* qw_bf    = (__bf16*)(ws + o); o += (size_t)DINNER * DMODEL / 2;
    __bf16* okw_bf   = (__bf16*)(ws + o); o += (size_t)DMODEL * DINNER / 2;
    __bf16* oqw_bf   = (__bf16*)(ws + o); o += (size_t)DMODEL * DINNER / 2;
    __bf16* keyn_bf  = (__bf16*)(ws + o); o += (size_t)BB * LL * DINNER / 2;
    __bf16* qn_bf    = (__bf16*)(ws + o); o += (size_t)BB * KK * DINNER / 2;

    float* out_key   = (float*)d_out;                            // (B,L,512)
    float* out_query = (float*)d_out + (size_t)BB * LL * DMODEL; // (B,K,512)

    // 0) pre-convert GEMM operands to bf16
    {
        struct { const float* src; __bf16* dst; int n; } cv[6] = {
            { in_key,       inkey_bf, BB * LL * DMODEL },
            { key_proj_w,   keyw_bf,  NPROJ * DMODEL },
            { in_query,     inq_bf,   BB * KK * DMODEL },
            { query_proj_w, qw_bf,    DINNER * DMODEL },
            { out_key_w,    okw_bf,   DMODEL * DINNER },
            { out_query_w,  oqw_bf,   DMODEL * DINNER },
        };
        for (int i = 0; i < 6; ++i) {
            const int n8 = cv[i].n / 8;
            f32_to_bf16_kernel<<<(n8 + 255) / 256, 256, 0, stream>>>(cv[i].src, cv[i].dst, n8);
        }
    }

    // 1) input projection: zxbcdt = in_key @ key_proj_w^T   (4096 x 2058 x 512)
    {
        dim3 grid((BB * LL) / 64, (NPROJ + 63) / 64);
        gemm_wmma_bf16<<<grid, 128, 0, stream>>>(inkey_bf, keyw_bf, zxbcdt,
                                                 BB * LL, NPROJ, DMODEL);
    }
    // 2) depthwise convs + SiLU (both directions)
    {
        const int total = BB * LL * CCONV;
        dwconv_silu_kernel<<<(total + 255) / 256, 256, 0, stream>>>(
            zxbcdt, key_conv_w, key_conv_b, key_conv_bw, key_conv_bb, xbc_f, xbc_b);
    }
    // 3) dt / b_base / c_base
    {
        const int total = BB * LL * KK;
        dt_base_kernel<<<(total + 255) / 256, 256, 0, stream>>>(
            dist, zxbcdt, bc_proj_w, dt_proj_w, dt_bias, dtb, bbase, cbase);
    }
    // 4) h0 projection: (B*K x 512) @ (512 x 1024)
    {
        dim3 grid((BB * KK) / 64, DINNER / 64);
        gemm_wmma_bf16<<<grid, 128, 0, stream>>>(inq_bf, qw_bf, h0lin,
                                                 BB * KK, DINNER, DMODEL);
    }
    // 5) bidirectional serial scan (2 dirs * B * H = 32 workgroups)
    issm_scan_kernel<<<2 * BB * HH, 256, 0, stream>>>(
        dtb, bbase, cbase, xbc_f, xbc_b, h0lin, A_log, D_param, y_f, y_b, hF);

    // 6) gate + RMSNorm (emits bf16)
    gate_rmsnorm_kernel<<<BB * LL, 256, 0, stream>>>(y_f, y_b, zxbcdt, key_norm_w, keyn_bf);

    // 7) out_key = keyn @ out_key_w^T  (4096 x 512 x 1024)
    {
        dim3 grid((BB * LL) / 64, DMODEL / 64);
        gemm_wmma_bf16<<<grid, 128, 0, stream>>>(keyn_bf, okw_bf, out_key,
                                                 BB * LL, DMODEL, DINNER);
    }
    // 8) q LayerNorm (emits bf16)
    qprep_kernel<<<BB * KK, 256, 0, stream>>>(hF, q_ln_w, q_ln_b, qn_bf);

    // 9) out_query = qn @ out_query_w^T  (128 x 512 x 1024)
    {
        dim3 grid((BB * KK) / 64, DMODEL / 64);
        gemm_wmma_bf16<<<grid, 128, 0, stream>>>(qn_bf, oqw_bf, out_query,
                                                 BB * KK, DMODEL, DINNER);
    }
}
